// HANModel_13554916786205
// MI455X (gfx1250) — matrix operand order repaired and checked by reading, not compiled, not run on hardware
//
#include <hip/hip_runtime.h>

typedef __attribute__((ext_vector_type(16))) __bf16 v16bf;
typedef __attribute__((ext_vector_type(8)))  float  v8f;

// ---------- helpers ----------
static __device__ __forceinline__ float lrelu(float x) { return x > 0.f ? x : 0.2f * x; }
// order-preserving float<->uint encoding for atomicMax on floats
static __device__ __forceinline__ unsigned encf(float f) {
  unsigned u = __float_as_uint(f);
  return (u >> 31) ? ~u : (u | 0x80000000u);
}
static __device__ __forceinline__ float decf(unsigned e) {
  return __uint_as_float((e >> 31) ? (e ^ 0x80000000u) : ~e);
}

// ---------- pack W[K,N] f32 -> bf16 B-fragment layout ----------
// WP[((kt*tilesN + nt)*32 + lane)*16 + i] = bf16(W[(kt*32 + (i<8 ? 8*(lane/16) + i : 16 + 8*(lane/16) + i - 8)) * N + nt*16 + lane%16])
__global__ void pack_w(const float* __restrict__ W, __bf16* __restrict__ WP,
                       int K, int N, int tilesK, int tilesN)
{
  long idx = (long)blockIdx.x * blockDim.x + threadIdx.x;
  long total = (long)tilesK * tilesN * 32 * 16;
  if (idx >= total) return;
  int i    = (int)(idx & 15);
  int lane = (int)((idx >> 4) & 31);
  long t   = idx >> 9;
  int nt = (int)(t % tilesN);
  int kt = (int)(t / tilesN);
  int g = lane >> 4, lm = lane & 15;
  int k = kt * 32 + ((i < 8) ? (8 * g + i) : (16 + 8 * g + (i - 8)));
  int n = nt * 16 + lm;
  float v = (k < K && n < N) ? W[(long)k * N + n] : 0.f;
  WP[idx] = (__bf16)v;
}

// ---------- WMMA GEMM:  D = X[M,K] @ W[K,N] + b[N] ----------
// K multiple of 32; tilesN must be a multiple of JN (compile-time N-tiles per wave).
// MODE 0: store D. MODE 1: atomicAdd per-column sums of tanh(D) into colsum[N].
template <int JN, int MODE>
__global__ __launch_bounds__(256) void gemm_wmma(
    const float* __restrict__ X, const __bf16* __restrict__ WP,
    const float* __restrict__ bias, float* __restrict__ Out,
    float* __restrict__ colsum, int M, int K, int N)
{
  const int lane = threadIdx.x & 31;
  const int wave = threadIdx.x >> 5;
  const int tilesN  = (N + 15) >> 4;
  const int tilesM  = (M + 15) >> 4;
  const int groupsN = tilesN / JN;                // exact by contract
  const long wid = (long)blockIdx.x * 8 + wave;
  if (wid >= (long)tilesM * groupsN) return;      // wave-uniform exit: EXEC stays full
  const int tm  = (int)(wid / groupsN);
  const int tn0 = (int)(wid % groupsN) * JN;
  const int g  = lane >> 4;
  const int lm = lane & 15;
  const int row  = tm * 16 + lm;
  const int rowc = (row < M) ? row : (M - 1);     // clamp: loads stay in-bounds, no EXEC games
  const bool rowOK = (row < M);

  v8f acc[JN];
#pragma unroll
  for (int j = 0; j < JN; ++j) acc[j] = (v8f){};

  // per-wave base pointers; inner loop advances by constant strides
  const float4* xp = (const float4*)X + (((long)rowc * K) >> 2);
  const __bf16* wb = WP + (((size_t)tn0) * 32 + lane) * 16;   // + kt*tilesN*512 per K-step
  const size_t ktStride = (size_t)tilesN * 512;               // bf16 elems per K-step
  const int kSteps = K >> 5;

  for (int kt = 0; kt < kSteps; ++kt) {
    // ---- preload all B fragments (constant 1024B tile stride -> clustered b128 loads) ----
    v16bf bfrag[JN];
#pragma unroll
    for (int j = 0; j < JN; ++j)
      bfrag[j] = *(const v16bf*)(wb + (size_t)j * 512);
    wb += ktStride;

    // ---- load + convert A fragment (two contiguous 8-float runs per lane) ----
    float4 a0 = xp[2 * g];                        // k = kk+8g   .. +3
    float4 a1 = xp[2 * g + 1];                    // k = kk+8g+4 .. +7
    float4 a2 = xp[4 + 2 * g];                    // k = kk+16+8g ..
    float4 a3 = xp[5 + 2 * g];
    xp += 8;                                      // 32 floats
    if (!rowOK) {
      a0 = make_float4(0.f, 0.f, 0.f, 0.f); a1 = a0; a2 = a0; a3 = a0;
    }
    v16bf a;
    a[0]  = (__bf16)a0.x; a[1]  = (__bf16)a0.y; a[2]  = (__bf16)a0.z; a[3]  = (__bf16)a0.w;
    a[4]  = (__bf16)a1.x; a[5]  = (__bf16)a1.y; a[6]  = (__bf16)a1.z; a[7]  = (__bf16)a1.w;
    a[8]  = (__bf16)a2.x; a[9]  = (__bf16)a2.y; a[10] = (__bf16)a2.z; a[11] = (__bf16)a2.w;
    a[12] = (__bf16)a3.x; a[13] = (__bf16)a3.y; a[14] = (__bf16)a3.z; a[15] = (__bf16)a3.w;

    // ---- 4 back-to-back WMMAs on distinct accumulator banks ----
#pragma unroll
    for (int j = 0; j < JN; ++j)
      acc[j] = __builtin_amdgcn_wmma_f32_16x16x32_bf16(
          false, a, false, bfrag[j], (short)0, acc[j], false, false);
  }

  const bool fullM = (tm * 16 + 16 <= M);         // wave-uniform
#pragma unroll
  for (int j = 0; j < JN; ++j) {
    const int tn = tn0 + j;
    const int col = tn * 16 + lm;
    const float bn = (col < N) ? bias[col] : 0.f;
    if (MODE == 0) {
      if (fullM && tn * 16 + 16 <= N) {           // interior tile: unguarded stores
#pragma unroll
        for (int r = 0; r < 8; ++r)
          Out[(long)(tm * 16 + r + 8 * g) * N + col] = acc[j][r] + bn;
      } else {
#pragma unroll
        for (int r = 0; r < 8; ++r) {
          int m = tm * 16 + r + 8 * g;
          if (m < M && col < N) Out[(long)m * N + col] = acc[j][r] + bn;
        }
      }
    } else {
      float s = 0.f;
#pragma unroll
      for (int r = 0; r < 8; ++r) {
        int m = tm * 16 + r + 8 * g;
        if (m < M) s += tanhf(acc[j][r] + bn);
      }
      if (col < N) atomicAdd(&colsum[col], s);
    }
  }
}

// ---------- tiny k_lin colsum for F==2 (layer 3) ----------
__global__ void klin_small(const float* __restrict__ X, const float* __restrict__ W,
                           const float* __restrict__ bias, float* __restrict__ colsum,
                           int M, int F)
{
  int n = blockIdx.x * blockDim.x + threadIdx.x;
  if (n >= M) return;
  for (int f = 0; f < F; ++f) {
    float s = bias[f];
    for (int k = 0; k < F; ++k) s += X[(long)n * F + k] * W[(long)k * F + f];
    atomicAdd(&colsum[f], tanhf(s));
  }
}

// ---------- fills ----------
__global__ void fill_f32(float* p, float v, long n) {
  long i = (long)blockIdx.x * blockDim.x + threadIdx.x;
  if (i < n) p[i] = v;
}
__global__ void fill_u32(unsigned* p, unsigned v, long n) {
  long i = (long)blockIdx.x * blockDim.x + threadIdx.x;
  if (i < n) p[i] = v;
}

// ---------- per-node attention dot: out[n,h] = sum_c feat[n,h*C+c]*avec[h*C+c] ----------
__global__ void node_dot(const float* __restrict__ feat, const float* __restrict__ avec,
                         float* __restrict__ out, int Nn, int H, int C)
{
  long idx = (long)blockIdx.x * blockDim.x + threadIdx.x;
  if (idx >= (long)Nn * H) return;
  int n = (int)(idx / H), h = (int)(idx % H);
  const int F = H * C;
  const float* f = feat + (long)n * F + h * C;
  const float* a = avec + h * C;
  float s = 0.f;
  for (int c = 0; c < C; ++c) s += f[c] * a[c];
  out[idx] = s;
}

// ---------- edge softmax passes ----------
__global__ void edge_max(const int* __restrict__ src, const int* __restrict__ dst,
                         const float* __restrict__ as, const float* __restrict__ ad,
                         unsigned* __restrict__ mx, int E, int H)
{
  long idx = (long)blockIdx.x * blockDim.x + threadIdx.x;
  if (idx >= (long)E * H) return;
  int e = (int)(idx / H), h = (int)(idx % H);
  int s = src[e], d = dst[e];
  float l = lrelu(as[(long)s * H + h] + ad[(long)d * H + h]);
  atomicMax(&mx[(long)d * H + h], encf(l));
}
__global__ void edge_sum(const int* __restrict__ src, const int* __restrict__ dst,
                         const float* __restrict__ as, const float* __restrict__ ad,
                         const unsigned* __restrict__ mx, float* __restrict__ sm, int E, int H)
{
  long idx = (long)blockIdx.x * blockDim.x + threadIdx.x;
  if (idx >= (long)E * H) return;
  int e = (int)(idx / H), h = (int)(idx % H);
  int s = src[e], d = dst[e];
  long dh = (long)d * H + h;
  float l = lrelu(as[(long)s * H + h] + ad[dh]);
  atomicAdd(&sm[dh], __expf(l - decf(mx[dh])));
}
__global__ void edge_scatter(const int* __restrict__ src, const int* __restrict__ dst,
                             const float* __restrict__ as, const float* __restrict__ ad,
                             const unsigned* __restrict__ mx, const float* __restrict__ sm,
                             const float* __restrict__ hsrc, float* __restrict__ agg,
                             int E, int H, int C)
{
  long idx = (long)blockIdx.x * blockDim.x + threadIdx.x;
  if (idx >= (long)E * H) return;
  int e = (int)(idx / H), h = (int)(idx % H);
  int s = src[e], d = dst[e];
  long dh = (long)d * H + h;
  const int F = H * C;
  float l = lrelu(as[(long)s * H + h] + ad[dh]);
  float w = __expf(l - decf(mx[dh])) / (sm[dh] + 1e-16f);
  const float* hp = hsrc + (long)s * F + h * C;
  float* ap = agg + (long)d * F + h * C;
  for (int c = 0; c < C; ++c) atomicAdd(&ap[c], hp[c] * w);
}

__global__ void relu_ip(float* p, long n) {
  long i = (long)blockIdx.x * blockDim.x + threadIdx.x;
  if (i < n) p[i] = fmaxf(p[i], 0.f);
}

// ---------- semantic attention score -> beta (2 metapaths) ----------
__global__ void score_beta(const float* __restrict__ colsum, const float* __restrict__ q,
                           float* __restrict__ beta, int F, float invN)
{
  float s0 = 0.f, s1 = 0.f;
  for (int f = 0; f < F; ++f) {
    s0 += colsum[f] * invN * q[f];
    s1 += colsum[F + f] * invN * q[f];
  }
  float m = fmaxf(s0, s1);
  float e0 = __expf(s0 - m), e1 = __expf(s1 - m);
  beta[0] = e0 / (e0 + e1);
  beta[1] = e1 / (e0 + e1);
}

__global__ void combine2(const float* __restrict__ a0, const float* __restrict__ a1,
                         const float* __restrict__ beta, float* __restrict__ out,
                         long n, int doElu)
{
  long i = (long)blockIdx.x * blockDim.x + threadIdx.x;
  if (i >= n) return;
  float v = beta[0] * a0[i] + beta[1] * a1[i];
  if (doElu) v = v > 0.f ? v : (__expf(v) - 1.f);
  out[i] = v;
}
__global__ void copy_act(const float* __restrict__ a, float* __restrict__ out, long n, int doElu) {
  long i = (long)blockIdx.x * blockDim.x + threadIdx.x;
  if (i >= n) return;
  float v = a[i];
  if (doElu) v = v > 0.f ? v : (__expf(v) - 1.f);
  out[i] = v;
}

// ---------- host ----------
static inline int blk(long n, int t) { return (int)((n + t - 1) / t); }

extern "C" void kernel_launch(void* const* d_in, const int* in_sizes, int n_in,
                              void* d_out, int out_size, void* d_ws, size_t ws_size,
                              hipStream_t stream)
{
  (void)in_sizes; (void)out_size; (void)ws_size;
  if (n_in < 58) return;

  const int NU = 100000, NH = 50000, NR = 5000;
  const size_t NT = 155000;     // feature arenas: URL@0, HAR@100000, Reg@150000
  const size_t NAGG = 255000;   // agg arena: UH@0[NH], HU@50000[NU], UR@150000[NR], RU@155000[NU]

  float* ws = (float*)d_ws;
  size_t off = 0;
  float*    HA   = ws + off; off += NT * 128;         // projected features h
  float*    RA   = ws + off; off += NT * 128;         // layer output / next input
  float*    AGG  = ws + off; off += NAGG * 128;       // per-metapath aggregates
  float*    AS   = ws + off; off += NAGG * 8;         // a_src dot per edge type
  float*    AD   = ws + off; off += NAGG * 8;         // a_dst dot per edge type
  unsigned* MAXB = (unsigned*)(ws + off); off += NAGG * 8;
  float*    SUMB = ws + off; off += NAGG * 8;
  float*    COLS = ws + off; off += 256;
  float*    BETA = ws + off; off += 8;
  __bf16*   WP   = (__bf16*)(ws + off); off += 16384; // 32768 bf16, packed weight tiles

  const float* xin[3] = { (const float*)d_in[0], (const float*)d_in[1], (const float*)d_in[2] };
  const int*   ei[4]  = { (const int*)d_in[3], (const int*)d_in[4],
                          (const int*)d_in[5], (const int*)d_in[6] };

  // edge types: 0=URL->HAR 1=HAR->URL 2=URL->Registrar 3=Registrar->URL
  const int Ecnt[4]   = { 400000, 400000, 100000, 100000 };
  const int srcOff[4] = { 0, 100000, 0, 150000 };       // feature-arena node offsets
  const int dstOff[4] = { 100000, 0, 150000, 0 };
  const int NsrcA[4]  = { NU, NH, NU, NR };
  const int NdstA[4]  = { NH, NU, NR, NU };
  const int asOff[4]  = { 0, 100000, 150000, 250000 };  // AS arena node offsets
  const int adOff[4]  = { 0, 50000, 150000, 155000 };   // AD/MAX/SUM arena node offsets
  const int aggOff[4] = { 0, 50000, 150000, 155000 };   // AGG arena node offsets
  const int aSrcIdx[4] = { 6, 4, 7, 5 };                // leaf index within layer
  const int aDstIdx[4] = { 2, 0, 3, 1 };

  for (int L = 0; L < 3; ++L) {
    const int H = (L < 2) ? 8 : 1;
    const int C = (L < 2) ? 16 : 2;
    const int F = H * C;
    const float* const* P = (const float* const*)(d_in + 7 + L * 17);

    // --- projections: h[t] = x[t] @ W_t + b_t ---
    const int Mt[3]   = { NU, NH, NR };
    const int Kt[3]   = { (L == 0) ? 128 : 128, (L == 0) ? 64 : 128, (L == 0) ? 32 : 128 };
    const int foff[3] = { 0, 100000, 150000 };
    const int widx[3] = { 14, 10, 12 };
    const int bidx[3] = { 15, 11, 13 };
    const int tilesN = (F + 15) / 16;                 // 8 (L<2) or 1 (L==2)
    for (int t = 0; t < 3; ++t) {
      const float* X = (L == 0) ? xin[t] : (RA + (size_t)foff[t] * 128);
      const int tilesK = (Kt[t] + 31) / 32;
      pack_w<<<blk((long)tilesK * tilesN * 512, 256), 256, 0, stream>>>(
          P[widx[t]], WP, Kt[t], F, tilesK, tilesN);
      float* OutP = HA + (size_t)foff[t] * F;
      if (L < 2) {
        long waves = (long)((Mt[t] + 15) / 16) * (tilesN / 4);
        gemm_wmma<4, 0><<<blk(waves, 8), 256, 0, stream>>>(
            X, WP, P[bidx[t]], OutP, nullptr, Mt[t], Kt[t], F);
      } else {
        long waves = (long)((Mt[t] + 15) / 16) * tilesN;
        gemm_wmma<1, 0><<<blk(waves, 8), 256, 0, stream>>>(
            X, WP, P[bidx[t]], OutP, nullptr, Mt[t], Kt[t], F);
      }
    }

    // --- init softmax / agg buffers ---
    fill_u32<<<blk((long)NAGG * H, 256), 256, 0, stream>>>(MAXB, 0x007FFFFFu, (long)NAGG * H); // enc(-inf)
    fill_f32<<<blk((long)NAGG * H, 256), 256, 0, stream>>>(SUMB, 0.f, (long)NAGG * H);
    fill_f32<<<blk((long)NAGG * F, 256), 256, 0, stream>>>(AGG, 0.f, (long)NAGG * F);
    fill_f32<<<1, 256, 0, stream>>>(COLS, 0.f, 2L * F);

    // --- per edge type: attention softmax + weighted scatter ---
    for (int e = 0; e < 4; ++e) {
      const int* srcp = ei[e];
      const int* dstp = ei[e] + Ecnt[e];
      float* asP = AS + (size_t)asOff[e] * H;
      float* adP = AD + (size_t)adOff[e] * H;
      unsigned* mxP = MAXB + (size_t)adOff[e] * H;
      float* smP = SUMB + (size_t)adOff[e] * H;
      node_dot<<<blk((long)NsrcA[e] * H, 256), 256, 0, stream>>>(
          HA + (size_t)srcOff[e] * F, P[aSrcIdx[e]], asP, NsrcA[e], H, C);
      node_dot<<<blk((long)NdstA[e] * H, 256), 256, 0, stream>>>(
          HA + (size_t)dstOff[e] * F, P[aDstIdx[e]], adP, NdstA[e], H, C);
      long eh = (long)Ecnt[e] * H;
      edge_max<<<blk(eh, 256), 256, 0, stream>>>(srcp, dstp, asP, adP, mxP, Ecnt[e], H);
      edge_sum<<<blk(eh, 256), 256, 0, stream>>>(srcp, dstp, asP, adP, mxP, smP, Ecnt[e], H);
      edge_scatter<<<blk(eh, 256), 256, 0, stream>>>(
          srcp, dstp, asP, adP, mxP, smP,
          HA + (size_t)srcOff[e] * F, AGG + (size_t)aggOff[e] * F, Ecnt[e], H, C);
    }
    relu_ip<<<blk((long)NAGG * F, 256), 256, 0, stream>>>(AGG, (long)NAGG * F);

    // --- semantic attention (only URL has 2 metapaths; others beta==1) ---
    if (L < 2) {
      const int tilesK2 = (F + 31) / 32, tilesN2 = (F + 15) / 16;  // 4, 8
      pack_w<<<blk((long)tilesK2 * tilesN2 * 512, 256), 256, 0, stream>>>(
          P[8], WP, F, F, tilesK2, tilesN2);
      long waves = (long)((NU + 15) / 16) * (tilesN2 / 4);
      gemm_wmma<4, 1><<<blk(waves, 8), 256, 0, stream>>>(           // HAR->URL stack
          AGG + (size_t)50000 * F, WP, P[9], nullptr, COLS, NU, F, F);
      gemm_wmma<4, 1><<<blk(waves, 8), 256, 0, stream>>>(           // Registrar->URL stack
          AGG + (size_t)155000 * F, WP, P[9], nullptr, COLS + F, NU, F, F);
    } else {
      klin_small<<<blk(NU, 256), 256, 0, stream>>>(
          AGG + (size_t)50000 * F, P[8], P[9], COLS, NU, F);
      klin_small<<<blk(NU, 256), 256, 0, stream>>>(
          AGG + (size_t)155000 * F, P[8], P[9], COLS + F, NU, F);
    }
    score_beta<<<1, 1, 0, stream>>>(COLS, P[16], BETA, F, 1.0f / NU);

    // --- combine + activation, write layer output ---
    const int doElu = (L < 2) ? 1 : 0;
    float* outURL = (L < 2) ? RA : (float*)d_out;
    combine2<<<blk((long)NU * F, 256), 256, 0, stream>>>(
        AGG + (size_t)50000 * F, AGG + (size_t)155000 * F, BETA, outURL, (long)NU * F, doElu);
    if (L < 2) {
      copy_act<<<blk((long)NH * F, 256), 256, 0, stream>>>(
          AGG, RA + (size_t)100000 * F, (long)NH * F, 1);                       // URL->HAR stack
      copy_act<<<blk((long)NR * F, 256), 256, 0, stream>>>(
          AGG + (size_t)150000 * F, RA + (size_t)150000 * F, (long)NR * F, 1);  // URL->Registrar stack
    }
  }
}
